// OuterProductMean_36060545417655
// MI455X (gfx1250) — matrix operand order repaired and checked by reading, not compiled, not run on hardware
//
#include <hip/hip_runtime.h>
#include <hip/hip_bf16.h>

// ---- CDNA5 (gfx1250) wave32 WMMA types ----
typedef _Float16 h8   __attribute__((ext_vector_type(8)));
typedef _Float16 v16h __attribute__((ext_vector_type(16)));
typedef float    v8f  __attribute__((ext_vector_type(8)));

union V16U { v16h v; h8 h[2]; };

#define S_  64
#define R_  192
#define C_  256
#define H_  32
#define CP_ 128

// =====================================================================
// Kernel 1: LayerNorm over C + left/right projections.
// grid = S_*R_ blocks, 64 threads. Output transposed f16:
//   LT[(r*H + h)*S + s], RT likewise  (so phase-2 lanes read rows contiguously)
// =====================================================================
__global__ void __launch_bounds__(64)
opm_ln_proj(const float* __restrict__ x, const float* __restrict__ gamma,
            const float* __restrict__ beta,
            const float* __restrict__ Wl, const float* __restrict__ bl,
            const float* __restrict__ Wr, const float* __restrict__ br,
            _Float16* __restrict__ LT, _Float16* __restrict__ RT)
{
    __shared__ float xs[C_];
    __shared__ float red[64][2];
    const int tid = threadIdx.x;
    const int bx  = blockIdx.x;
    const int s = bx / R_;
    const int r = bx % R_;

    const float4* xin = (const float4*)(x + (size_t)(s * R_ + r) * C_);
    float4 v = xin[tid];
    xs[tid * 4 + 0] = v.x; xs[tid * 4 + 1] = v.y;
    xs[tid * 4 + 2] = v.z; xs[tid * 4 + 3] = v.w;
    red[tid][0] = v.x + v.y + v.z + v.w;
    red[tid][1] = v.x * v.x + v.y * v.y + v.z * v.z + v.w * v.w;
    __syncthreads();
    for (int off = 32; off >= 1; off >>= 1) {
        if (tid < off) {
            red[tid][0] += red[tid + off][0];
            red[tid][1] += red[tid + off][1];
        }
        __syncthreads();
    }
    const float mu  = red[0][0] * (1.0f / C_);
    const float var = red[0][1] * (1.0f / C_) - mu * mu;
    const float rs  = rsqrtf(var + 1e-5f);
    #pragma unroll
    for (int t = 0; t < 4; ++t) {
        int c = tid * 4 + t;
        xs[c] = (xs[c] - mu) * rs * gamma[c] + beta[c];
    }
    __syncthreads();

    const int h = tid & 31;
    const float* W   = (tid < 32) ? Wl : Wr;
    const float* bb  = (tid < 32) ? bl : br;
    _Float16*    out = (tid < 32) ? LT : RT;
    float acc = 0.f;
    #pragma unroll 8
    for (int c = 0; c < C_; ++c) acc += xs[c] * W[c * H_ + h];
    acc += bb[h];
    out[(size_t)(r * H_ + h) * S_ + s] = (_Float16)acc;
}

// =====================================================================
// Kernel 2: pre-swizzle W_out (1024x128 f32) into f16 WMMA B-fragment
// lane order: ws[((nt*32 + kc)*32 + lane)*16 + t] = W[kc*32 + (lane>>4)*8 + (t&7) + (t>>3)*16][nt*16 + (lane&15)]
// so each lane loads its 16 B-halves as one contiguous 32B chunk.
// =====================================================================
__global__ void __launch_bounds__(256)
opm_swz_wout(const float* __restrict__ Wout, _Float16* __restrict__ Wswz)
{
    int gid  = blockIdx.x * 256 + threadIdx.x;   // 0..131071
    int t    = gid & 15;
    int lane = (gid >> 4) & 31;
    int kc   = (gid >> 9) & 31;
    int nt   = gid >> 14;
    int k = kc * 32 + ((lane >> 4) * 8) + (t & 7) + ((t >> 3) * 16);
    int p = nt * 16 + (lane & 15);
    Wswz[gid] = (_Float16)Wout[k * CP_ + p];
}

// =====================================================================
// Kernel 3: fused Gram + projection. 48x48 blocks, each a 4x4 (i,j) tile.
// 128 threads = 4 waves. Wave w handles i = i0+w, j = j0..j0+3 (4 pairs),
// then the projection N-tiles 2w, 2w+1 over M=16 pairs, K=1024.
// =====================================================================
__global__ void __launch_bounds__(128)
opm_main(const _Float16* __restrict__ LT, const _Float16* __restrict__ RT,
         const _Float16* __restrict__ Wswz, const float* __restrict__ bout,
         float* __restrict__ z)
{
    __shared__ __align__(16) _Float16 qT[4][32][72];      // per-wave q^T [d][s], padded rows
    __shared__ __align__(16) _Float16 outerS[16][1032];   // [pair][d*32+e], padded rows

    const int tid = threadIdx.x;
    const int w   = tid >> 5;
    const int l   = tid & 31;
    const int bi  = blockIdx.x / (R_ / 4);
    const int bj  = blockIdx.x % (R_ / 4);
    const int i0  = bi * 4, j0 = bj * 4;
    const int m     = l & 15;
    const int hi    = l >> 4;
    const int kbase = hi * 8;    // A-matrix K split per ISA layout
    const int rbase = hi * 16;   // unused; B reads use kbase, the same K split as A

    // ---------------- Gram phase: outer = q^T q / S per pair ----------------
    const int i = i0 + w;
    const h8* Lr = (const h8*)(LT + (size_t)(i * H_ + l) * S_);   // lane owns d = l
    #pragma unroll
    for (int q = 0; q < 4; ++q) {
        const int j  = j0 + q;
        const int pp = w * 4 + q;
        const h8* Rr = (const h8*)(RT + (size_t)(j * H_ + l) * S_);
        h8* qrow = (h8*)&qT[w][l][0];
        #pragma unroll
        for (int c = 0; c < 8; ++c) qrow[c] = Lr[c] * Rr[c];      // v_pk_mul_f16
        __builtin_amdgcn_wave_barrier();                          // order LDS RAW in-wave

        v8f a00 = {}, a01 = {}, a10 = {}, a11 = {};
        #pragma unroll
        for (int sc = 0; sc < 2; ++sc) {                          // K = s, chunks of 32
            V16U A0, A1, B0, B1;
            A0.h[0] = *(const h8*)&qT[w][m     ][sc * 32 + kbase];
            A0.h[1] = *(const h8*)&qT[w][m     ][sc * 32 + kbase + 16];
            A1.h[0] = *(const h8*)&qT[w][16 + m][sc * 32 + kbase];
            A1.h[1] = *(const h8*)&qT[w][16 + m][sc * 32 + kbase + 16];
            B0.h[0] = *(const h8*)&qT[w][m     ][sc * 32 + kbase];
            B0.h[1] = *(const h8*)&qT[w][m     ][sc * 32 + kbase + 16];
            B1.h[0] = *(const h8*)&qT[w][16 + m][sc * 32 + kbase];
            B1.h[1] = *(const h8*)&qT[w][16 + m][sc * 32 + kbase + 16];
            a00 = __builtin_amdgcn_wmma_f32_16x16x32_f16(false, A0.v, false, B0.v, (short)0, a00, false, false);
            a01 = __builtin_amdgcn_wmma_f32_16x16x32_f16(false, A0.v, false, B1.v, (short)0, a01, false, false);
            a10 = __builtin_amdgcn_wmma_f32_16x16x32_f16(false, A1.v, false, B0.v, (short)0, a10, false, false);
            a11 = __builtin_amdgcn_wmma_f32_16x16x32_f16(false, A1.v, false, B1.v, (short)0, a11, false, false);
        }
        const float sinv = 1.0f / (float)S_;
        #pragma unroll
        for (int r8 = 0; r8 < 8; ++r8) {                          // C layout: M = r8 + 8*hi, N = m
            int d0 = r8 + hi * 8;
            outerS[pp][(d0     ) * 32 + (m     )] = (_Float16)(a00[r8] * sinv);
            outerS[pp][(d0     ) * 32 + (16 + m)] = (_Float16)(a01[r8] * sinv);
            outerS[pp][(d0 + 16) * 32 + (m     )] = (_Float16)(a10[r8] * sinv);
            outerS[pp][(d0 + 16) * 32 + (16 + m)] = (_Float16)(a11[r8] * sinv);
        }
        __builtin_amdgcn_wave_barrier();
    }
    __syncthreads();

    // ---------------- Projection: z = outer(16x1024) @ W_out(1024x128) ------
    v8f z0 = {}, z1 = {};
    const _Float16* Wb0 = Wswz + (size_t)((2 * w    ) * 32) * 32 * 16;
    const _Float16* Wb1 = Wswz + (size_t)((2 * w + 1) * 32) * 32 * 16;
    #pragma unroll 2
    for (int kc = 0; kc < 32; ++kc) {
        V16U A, B0, B1;
        A.h[0] = *(const h8*)&outerS[m][kc * 32 + kbase];
        A.h[1] = *(const h8*)&outerS[m][kc * 32 + kbase + 16];
        const h8* bp0 = (const h8*)(Wb0 + (size_t)(kc * 32 + l) * 16);
        const h8* bp1 = (const h8*)(Wb1 + (size_t)(kc * 32 + l) * 16);
        __builtin_prefetch(bp0 + 64, 0, 1);                        // global_prefetch_b8
        B0.h[0] = bp0[0]; B0.h[1] = bp0[1];
        B1.h[0] = bp1[0]; B1.h[1] = bp1[1];
        z0 = __builtin_amdgcn_wmma_f32_16x16x32_f16(false, A.v, false, B0.v, (short)0, z0, false, false);
        z1 = __builtin_amdgcn_wmma_f32_16x16x32_f16(false, A.v, false, B1.v, (short)0, z1, false, false);
    }
    const float bo0 = bout[(2 * w    ) * 16 + m];
    const float bo1 = bout[(2 * w + 1) * 16 + m];
    #pragma unroll
    for (int r8 = 0; r8 < 8; ++r8) {
        int pp = r8 + hi * 8;                                      // pair = C row
        int ii = pp >> 2, jj = pp & 3;
        size_t base = ((size_t)((i0 + ii) * R_ + (j0 + jj))) * CP_;
        z[base + (2 * w    ) * 16 + m] = z0[r8] + bo0;
        z[base + (2 * w + 1) * 16 + m] = z1[r8] + bo1;
    }
}

// =====================================================================
extern "C" void kernel_launch(void* const* d_in, const int* in_sizes, int n_in,
                              void* d_out, int out_size, void* d_ws, size_t ws_size,
                              hipStream_t stream) {
    const float* x     = (const float*)d_in[0];
    const float* gamma = (const float*)d_in[1];
    const float* beta  = (const float*)d_in[2];
    const float* Wl    = (const float*)d_in[3];
    const float* bl    = (const float*)d_in[4];
    const float* Wr    = (const float*)d_in[5];
    const float* br    = (const float*)d_in[6];
    const float* Wout  = (const float*)d_in[7];
    const float* bout  = (const float*)d_in[8];
    float* z = (float*)d_out;

    // workspace layout (f16): LT | RT | Wswz  -> ~1.75 MB total
    _Float16* LT   = (_Float16*)d_ws;
    _Float16* RT   = LT + (size_t)R_ * H_ * S_;
    _Float16* Wswz = RT + (size_t)R_ * H_ * S_;

    opm_ln_proj<<<S_ * R_, 64, 0, stream>>>(x, gamma, beta, Wl, bl, Wr, br, LT, RT);
    opm_swz_wout<<<(H_ * H_ * CP_) / 256, 256, 0, stream>>>(Wout, Wswz);
    opm_main<<<(R_ / 4) * (R_ / 4), 128, 0, stream>>>(LT, RT, Wswz, bout, z);
}
